// FactorizedAttention_22514218565670
// MI455X (gfx1250) — compile-verified, matrix operand order checked by reading, not run on hardware
//
#include <hip/hip_runtime.h>
#include <hip/hip_bf16.h>

typedef __attribute__((ext_vector_type(16))) _Float16 v16h;
typedef __attribute__((ext_vector_type(8)))  _Float16 v8h;
typedef __attribute__((ext_vector_type(8)))  float    v8f;
typedef __attribute__((ext_vector_type(4))) unsigned int u32x4;
typedef __attribute__((ext_vector_type(8))) int          i32x8;
typedef __attribute__((ext_vector_type(4))) int          i32x4;

#define WMMA_F16(a, b, c) \
  __builtin_amdgcn_wmma_f32_16x16x32_f16(false, (a), false, (b), (short)0, (c), false, false)

#if defined(__has_builtin)
#  if __has_builtin(__builtin_amdgcn_tensor_load_to_lds) && \
      __has_builtin(__builtin_amdgcn_s_wait_tensorcnt)
#    define FA_HAVE_TDM 1
#  endif
#endif

#define DIMD 768

// ---------------------------------------------------------------------------
// Fragment helpers (layouts per ISA 7.12.2, wave32).
// A-matrix 16x32 f16: lanes 0-15 row=lane, K={0..7,16..23}; lanes 16-31 kb=8.
// B-matrix assumed mirror: lane n holds column n with same K striping, so a
// row-major operand loaded by rows gives B[k][n] = Op[n][k].
// ---------------------------------------------------------------------------
__device__ __forceinline__ v16h ld_frag(const _Float16* base, int ld, int row0, int k0) {
  const int lane = threadIdx.x & 31;
  const int r    = row0 + (lane & 15);
  const int kb   = (lane >> 4) << 3;
  const _Float16* p = base + (size_t)r * ld + k0 + kb;
  v8h lo = *(const v8h*)p;
  v8h hi = *(const v8h*)(p + 16);
  v16h f;
#pragma unroll
  for (int i = 0; i < 8; ++i) { f[i] = lo[i]; f[i + 8] = hi[i]; }
  return f;
}

__device__ __forceinline__ v16h ld_frag_scaled(const _Float16* base, int ld, int row0,
                                               int k0, float s) {
  const int lane = threadIdx.x & 31;
  const int r    = row0 + (lane & 15);
  const int kb   = (lane >> 4) << 3;
  const _Float16* p = base + (size_t)r * ld + k0 + kb;
  v8h lo = *(const v8h*)p;
  v8h hi = *(const v8h*)(p + 16);
  v16h f;
#pragma unroll
  for (int i = 0; i < 8; ++i) {
    f[i]     = (_Float16)((float)lo[i] * s);
    f[i + 8] = (_Float16)((float)hi[i] * s);
  }
  return f;
}

// Gather a 32(K) x 16(N) B-fragment from a ROW-major tile (i.e. use columns):
// B[k][n] = tile[k][col+n].
__device__ __forceinline__ v16h ld_frag_col(const _Float16* base, int ld, int col) {
  const int lane = threadIdx.x & 31;
  const int c    = col + (lane & 15);
  const int kb   = (lane >> 4) << 3;
  v16h f;
#pragma unroll
  for (int i = 0; i < 8; ++i) {
    f[i]     = base[(kb + i) * ld + c];
    f[i + 8] = base[(16 + kb + i) * ld + c];
  }
  return f;
}

// C/D 16x16 f32: lanes 0-15 (M=v, N=lane); lanes 16-31 (M=v+8, N=lane-16)
__device__ __forceinline__ void st_tile_f16(_Float16* base, int ld, int row0, int col0,
                                            const v8f& c) {
  const int lane = threadIdx.x & 31;
  const int n    = col0 + (lane & 15);
  const int r0   = row0 + ((lane >> 4) << 3);
#pragma unroll
  for (int v = 0; v < 8; ++v) base[(size_t)(r0 + v) * ld + n] = (_Float16)c[v];
}

__device__ __forceinline__ void st_tile_f32(float* base, int ld, int row0, int col0,
                                            const v8f& c) {
  const int lane = threadIdx.x & 31;
  const int n    = col0 + (lane & 15);
  const int r0   = row0 + ((lane >> 4) << 3);
#pragma unroll
  for (int v = 0; v < 8; ++v) base[(size_t)(r0 + v) * ld + n] = c[v];
}

// ---------------------------------------------------------------------------
// TDM: DMA a rows x cols f16 tile (row stride = stride elems) into LDS with
// 16B padding every 128B row -> LDS row stride 72 halves (bank-conflict free).
// ---------------------------------------------------------------------------
#ifdef FA_HAVE_TDM
__device__ __forceinline__ void tdm_load_tile_f16(unsigned lds_off, const _Float16* gsrc,
                                                  int rows, int cols, int stride) {
  const unsigned long long ga = (unsigned long long)(uintptr_t)gsrc;
  u32x4 g0;
  g0[0] = 1u;                                        // count=1 (valid user D#)
  g0[1] = lds_off;                                   // lds_addr (bytes)
  g0[2] = (unsigned)(ga & 0xffffffffull);            // global_addr[31:0]
  g0[3] = (unsigned)((ga >> 32) & 0x01ffffffull) | (2u << 30);  // [56:32] | type=2
  i32x8 g1;
  // data_size=1 (2B) | pad_enable | pad_interval=4 (128B) | pad_amount=3 (16B)
  g1[0] = 0x00010000 | (1 << 20) | (4 << 22) | (3 << 25);
  g1[1] = (cols & 0xffff) << 16;                     // tensor_dim0 lo
  g1[2] = ((cols >> 16) & 0xffff) | ((rows & 0xffff) << 16);   // td0 hi | td1 lo
  g1[3] = ((rows >> 16) & 0xffff) | ((cols & 0xffff) << 16);   // td1 hi | tile_dim0
  g1[4] = rows & 0xffff;                             // tile_dim1 (tile_dim2=0)
  g1[5] = stride;                                    // tensor_dim0_stride lo
  g1[6] = 0;
  g1[7] = 0;
  i32x4 z = {0, 0, 0, 0};
#if __clang_major__ >= 23
  i32x8 z8 = {0, 0, 0, 0, 0, 0, 0, 0};
  __builtin_amdgcn_tensor_load_to_lds(g0, g1, z, z, z8, 0);
#else
  __builtin_amdgcn_tensor_load_to_lds(g0, g1, z, z, 0);
#endif
}
#endif

// ---------------------------------------------------------------------------
// f32 -> f16 conversion
// ---------------------------------------------------------------------------
__global__ void fa_cvt_f16(const float* __restrict__ src, _Float16* __restrict__ dst, int n) {
  int i = blockIdx.x * blockDim.x + threadIdx.x;
  if (i < n) dst[i] = (_Float16)src[i];
}

// ---------------------------------------------------------------------------
// Y[8192 x 768] = Xh @ Wh^T (3 projections via blockIdx.z), f16 out.
// 128 threads = 4 waves; each wave computes a 32(M) x 64(N) strip:
// per K-step 6 fragment loads -> 8 WMMAs.
// ---------------------------------------------------------------------------
__global__ void fa_gemm_qkv(const _Float16* __restrict__ xh,
                            const _Float16* __restrict__ wh3,  // [3][768][768]
                            _Float16* __restrict__ qkv) {      // [3][8192][768]
  const int wave = threadIdx.x >> 5;
  const int m0   = blockIdx.y * 128 + wave * 32;
  const int n0   = blockIdx.x * 64;
  const _Float16* W = wh3 + (size_t)blockIdx.z * DIMD * DIMD;
  _Float16*       Y = qkv + (size_t)blockIdx.z * 8192 * DIMD;

  v8f c0[4] = {}, c1[4] = {};
#pragma unroll 2
  for (int k0 = 0; k0 < DIMD; k0 += 32) {
    if (k0 + 32 < DIMD) {
      __builtin_prefetch(xh + (size_t)m0 * DIMD + k0 + 32, 0, 0);        // global_prefetch_b8
      __builtin_prefetch(xh + (size_t)(m0 + 16) * DIMD + k0 + 32, 0, 0);
    }
    v16h a0 = ld_frag(xh, DIMD, m0,      k0);
    v16h a1 = ld_frag(xh, DIMD, m0 + 16, k0);
#pragma unroll
    for (int j = 0; j < 4; ++j) {
      v16h b = ld_frag(W, DIMD, n0 + 16 * j, k0);
      c0[j] = WMMA_F16(a0, b, c0[j]);
      c1[j] = WMMA_F16(a1, b, c1[j]);
    }
  }
#pragma unroll
  for (int j = 0; j < 4; ++j) {
    st_tile_f16(Y, DIMD, m0,      n0 + 16 * j, c0[j]);
    st_tile_f16(Y, DIMD, m0 + 16, n0 + 16 * j, c1[j]);
  }
}

// Final projection: out_f32[8192 x 768] = ctx @ Wo^T
__global__ void fa_gemm_out(const _Float16* __restrict__ ctx,
                            const _Float16* __restrict__ woh,
                            float* __restrict__ out) {
  const int wave = threadIdx.x >> 5;
  const int m0   = blockIdx.y * 128 + wave * 32;
  const int n0   = blockIdx.x * 64;

  v8f c0[4] = {}, c1[4] = {};
#pragma unroll 2
  for (int k0 = 0; k0 < DIMD; k0 += 32) {
    if (k0 + 32 < DIMD) {
      __builtin_prefetch(ctx + (size_t)m0 * DIMD + k0 + 32, 0, 0);
      __builtin_prefetch(ctx + (size_t)(m0 + 16) * DIMD + k0 + 32, 0, 0);
    }
    v16h a0 = ld_frag(ctx, DIMD, m0,      k0);
    v16h a1 = ld_frag(ctx, DIMD, m0 + 16, k0);
#pragma unroll
    for (int j = 0; j < 4; ++j) {
      v16h b = ld_frag(woh, DIMD, n0 + 16 * j, k0);
      c0[j] = WMMA_F16(a0, b, c0[j]);
      c1[j] = WMMA_F16(a1, b, c1[j]);
    }
  }
#pragma unroll
  for (int j = 0; j < 4; ++j) {
    st_tile_f32(out, DIMD, m0,      n0 + 16 * j, c0[j]);
    st_tile_f32(out, DIMD, m0 + 16, n0 + 16 * j, c1[j]);
  }
}

// ---------------------------------------------------------------------------
// Fused flash attention per (b, h): grid (N/64, H, B), 128 threads = 4 waves.
// Wave w owns query rows [64*bx + 16w, +16). Keys in tiles of 32; K and V
// tiles staged row-major into LDS by the Tensor Data Mover (padded rows),
// online softmax in f32 registers; P bounced through LDS to A-layout.
// ---------------------------------------------------------------------------
__global__ void fa_attention(const _Float16* __restrict__ Q,
                             const _Float16* __restrict__ K,
                             const _Float16* __restrict__ V,
                             _Float16* __restrict__ ctx) {
  constexpr int LDT = 72;   // padded tile row stride in halves (128B + 16B pad)
  constexpr int LDP = 40;
  __shared__ __align__(16) _Float16 Kt[32 * LDT];
  __shared__ __align__(16) _Float16 Vr[32 * LDT];
  __shared__ __align__(16) _Float16 Pb[4 * 16 * LDP];

  const int tid  = threadIdx.x;
  const int wave = tid >> 5;
  const int lane = tid & 31;
  const int n16  = lane & 15;
  const int hh   = lane >> 4;

  const int b    = blockIdx.z;
  const int h    = blockIdx.y;
  const int qrow = blockIdx.x * 64 + wave * 16;
  const int rowb = b * 2048;
  const int colh = h * 64;

  const float qscale = 0.03608439182435161f;  // 1/sqrt(768), full-D scale

  v16h qa0 = ld_frag_scaled(Q + colh, DIMD, rowb + qrow, 0,  qscale);
  v16h qa1 = ld_frag_scaled(Q + colh, DIMD, rowb + qrow, 32, qscale);

  v8f  o0 = {}, o1 = {}, o2 = {}, o3 = {};
  float mrow[8], lrow[8];
#pragma unroll
  for (int v = 0; v < 8; ++v) { mrow[v] = -1e30f; lrow[v] = 0.0f; }

  _Float16* pw = Pb + wave * 16 * LDP;
#ifdef FA_HAVE_TDM
  const unsigned kt_off = (unsigned)(uintptr_t)Kt;
  const unsigned vr_off = (unsigned)(uintptr_t)Vr;
#endif

  for (int kt = 0; kt < 2048; kt += 32) {
    const _Float16* kg = K + (size_t)(rowb + kt) * DIMD + colh;
    const _Float16* vg = V + (size_t)(rowb + kt) * DIMD + colh;

#ifdef FA_HAVE_TDM
    if (wave == 0) {
      tdm_load_tile_f16(kt_off, kg, 32, 64, DIMD);
      tdm_load_tile_f16(vr_off, vg, 32, 64, DIMD);
      __builtin_amdgcn_s_wait_tensorcnt(0);
    }
#else
    for (int i = tid; i < 256; i += 128) {
      const int r = i >> 3, cs = (i & 7) << 3;
      *(v8h*)(Kt + r * LDT + cs) = *(const v8h*)(kg + (size_t)r * DIMD + cs);
      *(v8h*)(Vr + r * LDT + cs) = *(const v8h*)(vg + (size_t)r * DIMD + cs);
    }
#endif
    __syncthreads();

    // ---- S = Q K^T : two 16x16 C tiles over 32 keys ----
    v8f s0 = {}, s1 = {};
    {
      v16h bk;
      bk = ld_frag(Kt, LDT,  0,  0); s0 = WMMA_F16(qa0, bk, s0);
      bk = ld_frag(Kt, LDT,  0, 32); s0 = WMMA_F16(qa1, bk, s0);
      bk = ld_frag(Kt, LDT, 16,  0); s1 = WMMA_F16(qa0, bk, s1);
      bk = ld_frag(Kt, LDT, 16, 32); s1 = WMMA_F16(qa1, bk, s1);
    }

    // ---- online softmax (row reductions inside each 16-lane half) ----
#pragma unroll
    for (int v = 0; v < 8; ++v) {
      float mx = fmaxf(s0[v], s1[v]);
      mx = fmaxf(mx, __shfl_xor(mx, 1, 32));
      mx = fmaxf(mx, __shfl_xor(mx, 2, 32));
      mx = fmaxf(mx, __shfl_xor(mx, 4, 32));
      mx = fmaxf(mx, __shfl_xor(mx, 8, 32));
      const float mnew  = fmaxf(mrow[v], mx);
      const float alpha = __expf(mrow[v] - mnew);
      const float p0 = __expf(s0[v] - mnew);
      const float p1 = __expf(s1[v] - mnew);
      float rs = p0 + p1;
      rs += __shfl_xor(rs, 1, 32);
      rs += __shfl_xor(rs, 2, 32);
      rs += __shfl_xor(rs, 4, 32);
      rs += __shfl_xor(rs, 8, 32);
      lrow[v] = lrow[v] * alpha + rs;
      mrow[v] = mnew;
      o0[v] *= alpha; o1[v] *= alpha; o2[v] *= alpha; o3[v] *= alpha;
      s0[v] = p0; s1[v] = p1;
    }

    // ---- bounce P through LDS: C-layout f32 -> A-layout f16 (16x32) ----
#pragma unroll
    for (int v = 0; v < 8; ++v) {
      pw[(v + 8 * hh) * LDP + n16]      = (_Float16)s0[v];
      pw[(v + 8 * hh) * LDP + 16 + n16] = (_Float16)s1[v];
    }
    __syncthreads();

    v16h pa = ld_frag(pw, LDP, 0, 0);

    // ---- O += P V : B-frags gathered column-wise from row-major V tile ----
    {
      v16h bv;
      bv = ld_frag_col(Vr, LDT,  0); o0 = WMMA_F16(pa, bv, o0);
      bv = ld_frag_col(Vr, LDT, 16); o1 = WMMA_F16(pa, bv, o1);
      bv = ld_frag_col(Vr, LDT, 32); o2 = WMMA_F16(pa, bv, o2);
      bv = ld_frag_col(Vr, LDT, 48); o3 = WMMA_F16(pa, bv, o3);
    }
    __syncthreads();
  }

  // ---- epilogue: ctx = O / l ----
  const int r0 = rowb + qrow + 8 * hh;
#pragma unroll
  for (int v = 0; v < 8; ++v) {
    const float inv = 1.0f / lrow[v];
    ctx[(size_t)(r0 + v) * DIMD + colh +  0 + n16] = (_Float16)(o0[v] * inv);
    ctx[(size_t)(r0 + v) * DIMD + colh + 16 + n16] = (_Float16)(o1[v] * inv);
    ctx[(size_t)(r0 + v) * DIMD + colh + 32 + n16] = (_Float16)(o2[v] * inv);
    ctx[(size_t)(r0 + v) * DIMD + colh + 48 + n16] = (_Float16)(o3[v] * inv);
  }
}

// ---------------------------------------------------------------------------
// Launcher
// ---------------------------------------------------------------------------
extern "C" void kernel_launch(void* const* d_in, const int* in_sizes, int n_in,
                              void* d_out, int out_size, void* d_ws, size_t ws_size,
                              hipStream_t stream) {
  const float* x  = (const float*)d_in[0];
  const float* Wq = (const float*)d_in[1];
  const float* Wk = (const float*)d_in[2];
  const float* Wv = (const float*)d_in[3];
  const float* Wo = (const float*)d_in[4];
  float* out = (float*)d_out;

  const size_t NX = 8192ull * 768ull;
  const size_t NW = 768ull * 768ull;

  _Float16* xh  = (_Float16*)d_ws;
  _Float16* wh3 = xh  + NX;
  _Float16* woh = wh3 + 3 * NW;
  _Float16* qkv = woh + NW;
  _Float16* ctx = qkv + 3 * NX;

  fa_cvt_f16<<<(int)((NX + 255) / 256), 256, 0, stream>>>(x, xh, (int)NX);
  fa_cvt_f16<<<(int)((NW + 255) / 256), 256, 0, stream>>>(Wq, wh3,          (int)NW);
  fa_cvt_f16<<<(int)((NW + 255) / 256), 256, 0, stream>>>(Wk, wh3 + NW,     (int)NW);
  fa_cvt_f16<<<(int)((NW + 255) / 256), 256, 0, stream>>>(Wv, wh3 + 2 * NW, (int)NW);
  fa_cvt_f16<<<(int)((NW + 255) / 256), 256, 0, stream>>>(Wo, woh,          (int)NW);

  fa_gemm_qkv<<<dim3(12, 64, 3), 128, 0, stream>>>(xh, wh3, qkv);

  fa_attention<<<dim3(32, 12, 4), 128, 0, stream>>>(qkv, qkv + NX, qkv + 2 * NX, ctx);

  fa_gemm_out<<<dim3(12, 64, 1), 128, 0, stream>>>(ctx, woh, out);
}
